// DiffusionNetBlock_42880953483698
// MI455X (gfx1250) — matrix operand order repaired and checked by reading, not compiled, not run on hardware
//
#include <hip/hip_runtime.h>
#include <hip/hip_bf16.h>

typedef unsigned short u16;
typedef unsigned int   u32;
typedef __attribute__((ext_vector_type(16))) __bf16 v16bf;
typedef __attribute__((ext_vector_type(8)))  float  v8f;

#define NB 4
#define NN 20000
#define PP 128
#define KK 128
#define EE 160000

// ---------- helpers ----------

__device__ __forceinline__ u16 f2bf(float f) {
  u32 u = __float_as_uint(f);
  u32 r = u + 0x7FFFu + ((u >> 16) & 1u);   // round-to-nearest-even
  return (u16)(r >> 16);
}
__device__ __forceinline__ u32 pack2(float a, float b) {
  return (u32)f2bf(a) | ((u32)f2bf(b) << 16);
}

// A operand fragment (16x32 bf16, M x K), source row-major with stride lda.
// ISA layout: lanes 0-15 -> M=r, K pairs {2j,2j+1} in VGPR j (j<4: K=8*hf+2j, j>=4: K=16+8*hf+2(j-4))
__device__ __forceinline__ void load_a_bf(const u16* A, int lda, int m0, int k0,
                                          int lane, u32 u[8]) {
  int hf = (lane >> 4) & 1;
  int r  = lane & 15;
  const u16* p = A + (size_t)(m0 + r) * lda + (k0 + 8 * hf);
#pragma unroll
  for (int j = 0; j < 4; ++j) u[j] = *(const u32*)(p + 2 * j);
  p += 16;
#pragma unroll
  for (int j = 0; j < 4; ++j) u[4 + j] = *(const u32*)(p + 2 * j);
}

// Same but converting on the fly from f32 source.
__device__ __forceinline__ void load_a_f32(const float* A, int lda, int m0, int k0,
                                           int lane, u32 u[8]) {
  int hf = (lane >> 4) & 1;
  int r  = lane & 15;
  const float* p = A + (size_t)(m0 + r) * lda + (k0 + 8 * hf);
#pragma unroll
  for (int j = 0; j < 4; ++j) u[j] = pack2(p[2 * j], p[2 * j + 1]);
  p += 16;
#pragma unroll
  for (int j = 0; j < 4; ++j) u[4 + j] = pack2(p[2 * j], p[2 * j + 1]);
}

// B operand fragment (32x16 bf16, K x N), element (k, n) = BT[(n0+n)*ldb + k0 + k]
// i.e. BT is the transposed B ("weights row-major": out-col major, K contiguous).
// ISA layout: VGPR j holds K pair (j<4: K=8*hf+2j, j>=4: K=16+8*hf+2(j-4)) at column n = lane&15.
__device__ __forceinline__ void load_b_bf(const u16* BT, int ldb, int n0, int k0,
                                          int lane, u32 u[8]) {
  int hf = (lane >> 4) & 1;
  int n  = lane & 15;
  const u16* p = BT + (size_t)(n0 + n) * ldb + (k0 + 8 * hf);
#pragma unroll
  for (int j = 0; j < 4; ++j) u[j] = *(const u32*)(p + 2 * j);
  p += 16;
#pragma unroll
  for (int j = 0; j < 4; ++j) u[4 + j] = *(const u32*)(p + 2 * j);
}

__device__ __forceinline__ v8f wmma_bf16(const u32 a[8], const u32 b[8], v8f c) {
  union { u32 u[8]; v16bf v; } A, B;
#pragma unroll
  for (int j = 0; j < 8; ++j) { A.u[j] = a[j]; B.u[j] = b[j]; }
  return __builtin_amdgcn_wmma_f32_16x16x32_bf16(false, A.v, false, B.v,
                                                 (short)0, c, false, false);
}

// ---------- utility kernels ----------

__global__ void k_zero_f4(float4* p, long n4) {
  long i = blockIdx.x * (long)blockDim.x + threadIdx.x;
  if (i < n4) p[i] = make_float4(0.f, 0.f, 0.f, 0.f);
}

__global__ void k_f32_to_bf16(const float* src, u16* dst, int n) {
  int i = blockIdx.x * blockDim.x + threadIdx.x;
  if (i < n) dst[i] = f2bf(src[i]);
}

// bf16 copies of x_in, evecs; transposed bf16 of evecs (K x N) and x*area (P x N)
__global__ void k_prep(const float* __restrict__ x_in, const float* __restrict__ va,
                       const float* __restrict__ evecs,
                       u16* xin_bf, u16* xaT, u16* evecs_bf, u16* evecsT) {
  int idx = blockIdx.x * blockDim.x + threadIdx.x;   // over NB*NN
  if (idx >= NB * NN) return;
  int b = idx / NN, n = idx % NN;
  float area = va[idx];
  const float* xr = x_in + (size_t)idx * PP;
  u16* xo   = xin_bf + (size_t)idx * PP;
  u16* xaTb = xaT + (size_t)b * PP * NN;
  for (int p = 0; p < PP; ++p) {
    float x = xr[p];
    xo[p] = f2bf(x);
    xaTb[(size_t)p * NN + n] = f2bf(x * area);
  }
  const float* er = evecs + (size_t)idx * KK;
  u16* eo  = evecs_bf + (size_t)idx * KK;
  u16* eTb = evecsT + (size_t)b * KK * NN;
  for (int k = 0; k < KK; ++k) {
    float e = er[k];
    eo[k] = f2bf(e);
    eTb[(size_t)k * NN + n] = f2bf(e);
  }
}

// ---------- GEMM 1: x_spec(K,P) = evecs^T @ (x*area), split-N + f32 atomics ----------

__global__ void __launch_bounds__(256) k_gemm1(const u16* __restrict__ evecsT,
                                               const u16* __restrict__ xaT,
                                               float* __restrict__ xspec) {
  int b  = blockIdx.y;
  int n0 = blockIdx.x * 512;
  int nend = n0 + 512; if (nend > NN) nend = NN;
  const u16* At = evecsT + (size_t)b * KK * NN;   // K x N
  const u16* Bt = xaT    + (size_t)b * PP * NN;   // P x N (== B^T of N x P)
  int wave = threadIdx.x >> 5, lane = threadIdx.x & 31;
  int m0 = wave * 16;
  v8f acc[8];
#pragma unroll
  for (int t = 0; t < 8; ++t)
#pragma unroll
    for (int v = 0; v < 8; ++v) acc[t][v] = 0.f;

  for (int k = n0; k < nend; k += 32) {
    u32 a[8]; load_a_bf(At, NN, m0, k, lane, a);
#pragma unroll
    for (int t = 0; t < 8; ++t) {
      u32 bb[8]; load_b_bf(Bt, NN, t * 16, k, lane, bb);
      acc[t] = wmma_bf16(a, bb, acc[t]);
    }
  }
  int hf = (lane >> 4) & 1, n = lane & 15;
  float* out = xspec + (size_t)b * KK * PP;
#pragma unroll
  for (int t = 0; t < 8; ++t)
#pragma unroll
    for (int v = 0; v < 8; ++v)
      atomicAdd(&out[(m0 + v + 8 * hf) * PP + t * 16 + n], acc[t][v]);
}

// ---------- spectral heat scaling: yT(P,K) = bf16(exp(-eval*t) * x_spec) ----------

__global__ void k_scale(const float* __restrict__ evals, const float* __restrict__ times,
                        const float* __restrict__ xspec, u16* __restrict__ yT) {
  int idx = blockIdx.x * blockDim.x + threadIdx.x;   // over NB*KK*PP
  if (idx >= NB * KK * PP) return;
  int b = idx / (KK * PP);
  int k = (idx / PP) % KK;
  int p = idx % PP;
  float t = times[p]; t = t > 1e-8f ? t : 1e-8f;
  float s = __expf(-evals[b * KK + k] * t);
  yT[((size_t)b * PP + p) * KK + k] = f2bf(s * xspec[idx]);
}

// ---------- GEMM 2: x_diffuse(N,P) = evecs @ y ----------

__global__ void __launch_bounds__(256) k_gemm2(const u16* __restrict__ evecs_bf,
                                               const u16* __restrict__ yT,
                                               float* __restrict__ xd,
                                               u16* __restrict__ xd_bf) {
  int b = blockIdx.y;
  int wave = threadIdx.x >> 5, lane = threadIdx.x & 31;
  int tile = blockIdx.x * 8 + wave;
  if (tile * 16 >= NN) return;                     // wave-uniform guard
  int m0 = tile * 16;
  const u16* A  = evecs_bf + (size_t)b * NN * KK;  // N x K
  const u16* Bt = yT       + (size_t)b * PP * KK;  // P x K (== B^T of K x P)
  v8f acc[8];
#pragma unroll
  for (int t = 0; t < 8; ++t)
#pragma unroll
    for (int v = 0; v < 8; ++v) acc[t][v] = 0.f;

#pragma unroll
  for (int k = 0; k < KK; k += 32) {
    u32 a[8]; load_a_bf(A, KK, m0, k, lane, a);
#pragma unroll
    for (int t = 0; t < 8; ++t) {
      u32 bb[8]; load_b_bf(Bt, KK, t * 16, k, lane, bb);
      acc[t] = wmma_bf16(a, bb, acc[t]);
    }
  }
  int hf = (lane >> 4) & 1, n = lane & 15;
  float* xdb = xd + (size_t)b * NN * PP;
  u16* xdbf = xd_bf + (size_t)b * NN * PP;
#pragma unroll
  for (int t = 0; t < 8; ++t)
#pragma unroll
    for (int v = 0; v < 8; ++v) {
      size_t o = (size_t)(m0 + v + 8 * hf) * PP + t * 16 + n;
      float f = acc[t][v];
      xdb[o] = f;
      xdbf[o] = f2bf(f);
    }
}

// ---------- SpMM: gx/gy scatter (one thread per edge x 4 channels) ----------

__global__ void __launch_bounds__(256) k_spmm(const int* __restrict__ rows,
                                              const int* __restrict__ cols,
                                              const float* __restrict__ vx,
                                              const float* __restrict__ vy,
                                              const float* __restrict__ xd,
                                              float* __restrict__ gx,
                                              float* __restrict__ gy) {
  long tid = blockIdx.x * 256L + threadIdx.x;      // over NB*EE*32
  if (tid >= (long)NB * EE * 32) return;
  int p4 = (int)(tid & 31);
  long e = tid >> 5;                               // flat (b, edge)
  int b = (int)(e / EE);
  int r = rows[e], c = cols[e];
  float wx = vx[e], wy = vy[e];
  const float4 x = ((const float4*)(xd + ((size_t)b * NN + c) * PP))[p4];
  float* gxr = gx + ((size_t)b * NN + r) * PP + p4 * 4;
  float* gyr = gy + ((size_t)b * NN + r) * PP + p4 * 4;
  atomicAdd(gxr + 0, wx * x.x); atomicAdd(gxr + 1, wx * x.y);
  atomicAdd(gxr + 2, wx * x.z); atomicAdd(gxr + 3, wx * x.w);
  atomicAdd(gyr + 0, wy * x.x); atomicAdd(gyr + 1, wy * x.y);
  atomicAdd(gyr + 2, wy * x.z); atomicAdd(gyr + 3, wy * x.w);
}

// ---------- fused tail: gradient features + 3-layer MLP + residual ----------

__global__ void __launch_bounds__(256) k_tail(
    const float* __restrict__ gx, const float* __restrict__ gy,
    const u16* __restrict__ Bre, const u16* __restrict__ Bim,
    const u16* __restrict__ xin_bf, const u16* __restrict__ xd_bf,
    const u16* __restrict__ W1, const float* __restrict__ b1,
    const u16* __restrict__ W2, const float* __restrict__ b2,
    const u16* __restrict__ W3, const float* __restrict__ b3,
    const float* __restrict__ x_in, float* __restrict__ out) {
  __shared__ u16 lds[8][16 * 128];                 // per-wave C->A layout staging (32 KB)
  int b = blockIdx.y;
  int wave = threadIdx.x >> 5, lane = threadIdx.x & 31;
  int tile = blockIdx.x * 8 + wave;
  if (tile * 16 >= NN) return;                     // wave-uniform guard
  int m0 = tile * 16;
  int hf = (lane >> 4) & 1, nn = lane & 15;
  const float* gxb = gx + (size_t)b * NN * PP;
  const float* gyb = gy + (size_t)b * NN * PP;
  u16* my = lds[wave];

  // -- spatial gradient: gBr = gx @ Bre^T, gBi = gy @ Bim^T --
  v8f ar[8], ai[8];
#pragma unroll
  for (int t = 0; t < 8; ++t)
#pragma unroll
    for (int v = 0; v < 8; ++v) { ar[t][v] = 0.f; ai[t][v] = 0.f; }
#pragma unroll
  for (int k = 0; k < PP; k += 32) {
    u32 ax[8]; load_a_f32(gxb, PP, m0, k, lane, ax);
    u32 ay[8]; load_a_f32(gyb, PP, m0, k, lane, ay);
#pragma unroll
    for (int t = 0; t < 8; ++t) {
      u32 br[8]; load_b_bf(Bre, PP, t * 16, k, lane, br);
      ar[t] = wmma_bf16(ax, br, ar[t]);
      u32 bi[8]; load_b_bf(Bim, PP, t * 16, k, lane, bi);
      ai[t] = wmma_bf16(ay, bi, ai[t]);
    }
  }
  // xg = tanh(gx*gBr + gy*gBi)  -> LDS as bf16 (A-operand layout source)
#pragma unroll
  for (int t = 0; t < 8; ++t)
#pragma unroll
    for (int v = 0; v < 8; ++v) {
      int row = v + 8 * hf, col = t * 16 + nn;
      size_t g = (size_t)(m0 + row) * PP + col;
      float f = tanhf(gxb[g] * ar[t][v] + gyb[g] * ai[t][v]);
      my[row * 128 + col] = f2bf(f);
    }

  // -- layer 1: relu([x_in | x_diffuse | xg] @ W1^T + b1) --
  v8f h[8];
#pragma unroll
  for (int t = 0; t < 8; ++t)
#pragma unroll
    for (int v = 0; v < 8; ++v) h[t][v] = 0.f;
  const u16* A0 = xin_bf + (size_t)b * NN * PP;
  const u16* A1 = xd_bf  + (size_t)b * NN * PP;
#pragma unroll
  for (int k = 0; k < PP; k += 32) {
    u32 a[8]; load_a_bf(A0, PP, m0, k, lane, a);
#pragma unroll
    for (int t = 0; t < 8; ++t) {
      u32 bb[8]; load_b_bf(W1, 384, t * 16, k, lane, bb);
      h[t] = wmma_bf16(a, bb, h[t]);
    }
  }
#pragma unroll
  for (int k = 0; k < PP; k += 32) {
    u32 a[8]; load_a_bf(A1, PP, m0, k, lane, a);
#pragma unroll
    for (int t = 0; t < 8; ++t) {
      u32 bb[8]; load_b_bf(W1, 384, t * 16, 128 + k, lane, bb);
      h[t] = wmma_bf16(a, bb, h[t]);
    }
  }
#pragma unroll
  for (int k = 0; k < PP; k += 32) {
    u32 a[8]; load_a_bf(my, 128, 0, k, lane, a);
#pragma unroll
    for (int t = 0; t < 8; ++t) {
      u32 bb[8]; load_b_bf(W1, 384, t * 16, 256 + k, lane, bb);
      h[t] = wmma_bf16(a, bb, h[t]);
    }
  }
#pragma unroll
  for (int t = 0; t < 8; ++t)
#pragma unroll
    for (int v = 0; v < 8; ++v) {
      int row = v + 8 * hf, col = t * 16 + nn;
      my[row * 128 + col] = f2bf(fmaxf(h[t][v] + b1[col], 0.f));
    }

  // -- layer 2: relu(h1 @ W2^T + b2) --
  v8f h2[8];
#pragma unroll
  for (int t = 0; t < 8; ++t)
#pragma unroll
    for (int v = 0; v < 8; ++v) h2[t][v] = 0.f;
#pragma unroll
  for (int k = 0; k < PP; k += 32) {
    u32 a[8]; load_a_bf(my, 128, 0, k, lane, a);
#pragma unroll
    for (int t = 0; t < 8; ++t) {
      u32 bb[8]; load_b_bf(W2, 128, t * 16, k, lane, bb);
      h2[t] = wmma_bf16(a, bb, h2[t]);
    }
  }
#pragma unroll
  for (int t = 0; t < 8; ++t)
#pragma unroll
    for (int v = 0; v < 8; ++v) {
      int row = v + 8 * hf, col = t * 16 + nn;
      my[row * 128 + col] = f2bf(fmaxf(h2[t][v] + b2[col], 0.f));
    }

  // -- layer 3 + residual: out = h2 @ W3^T + b3 + x_in --
  v8f h3[8];
#pragma unroll
  for (int t = 0; t < 8; ++t)
#pragma unroll
    for (int v = 0; v < 8; ++v) h3[t][v] = 0.f;
#pragma unroll
  for (int k = 0; k < PP; k += 32) {
    u32 a[8]; load_a_bf(my, 128, 0, k, lane, a);
#pragma unroll
    for (int t = 0; t < 8; ++t) {
      u32 bb[8]; load_b_bf(W3, 128, t * 16, k, lane, bb);
      h3[t] = wmma_bf16(a, bb, h3[t]);
    }
  }
  const float* xinb = x_in + (size_t)b * NN * PP;
  float* ob = out + (size_t)b * NN * PP;
#pragma unroll
  for (int t = 0; t < 8; ++t)
#pragma unroll
    for (int v = 0; v < 8; ++v) {
      int row = v + 8 * hf, col = t * 16 + nn;
      size_t g = (size_t)(m0 + row) * PP + col;
      ob[g] = h3[t][v] + b3[col] + xinb[g];
    }
}

// ---------- launch ----------

extern "C" void kernel_launch(void* const* d_in, const int* in_sizes, int n_in,
                              void* d_out, int out_size, void* d_ws, size_t ws_size,
                              hipStream_t stream) {
  (void)in_sizes; (void)n_in; (void)out_size; (void)ws_size;
  const float* x_in  = (const float*)d_in[0];
  const float* va    = (const float*)d_in[1];
  const float* evals = (const float*)d_in[2];
  const float* evecs = (const float*)d_in[3];
  const float* gxv   = (const float*)d_in[4];
  const float* gyv   = (const float*)d_in[5];
  const int*   grows = (const int*)d_in[6];
  const int*   gcols = (const int*)d_in[7];
  const float* times = (const float*)d_in[8];
  const float* Bre   = (const float*)d_in[9];
  const float* Bim   = (const float*)d_in[10];
  const float* W1    = (const float*)d_in[11];
  const float* b1    = (const float*)d_in[12];
  const float* W2    = (const float*)d_in[13];
  const float* b2    = (const float*)d_in[14];
  const float* W3    = (const float*)d_in[15];
  const float* b3    = (const float*)d_in[16];
  float* out = (float*)d_out;

  char* w = (char*)d_ws;
  auto alloc = [&](size_t bytes) -> char* {
    char* p = w; w += (bytes + 255) & ~(size_t)255; return p;
  };
  u16*   evecs_bf  = (u16*)alloc((size_t)NB * NN * KK * 2);
  u16*   evecsT_bf = (u16*)alloc((size_t)NB * KK * NN * 2);
  u16*   xaT_bf    = (u16*)alloc((size_t)NB * PP * NN * 2);
  u16*   xin_bf    = (u16*)alloc((size_t)NB * NN * PP * 2);
  float* xd        = (float*)alloc((size_t)NB * NN * PP * 4);
  u16*   xd_bf     = (u16*)alloc((size_t)NB * NN * PP * 2);
  float* gx        = (float*)alloc((size_t)NB * NN * PP * 4);
  float* gy        = (float*)alloc((size_t)NB * NN * PP * 4);
  float* xspec     = (float*)alloc((size_t)NB * KK * PP * 4);
  u16*   yT_bf     = (u16*)alloc((size_t)NB * PP * KK * 2);
  u16*   Bre_bf    = (u16*)alloc(16384 * 2);
  u16*   Bim_bf    = (u16*)alloc(16384 * 2);
  u16*   W1_bf     = (u16*)alloc(49152 * 2);
  u16*   W2_bf     = (u16*)alloc(16384 * 2);
  u16*   W3_bf     = (u16*)alloc(16384 * 2);

  // zero f32 accumulators (ws is poisoned, not re-zeroed between replays)
  {
    long n4 = ((long)NB * NN * PP) / 4;
    int blocks = (int)((n4 + 255) / 256);
    k_zero_f4<<<blocks, 256, 0, stream>>>((float4*)gx, n4);
    k_zero_f4<<<blocks, 256, 0, stream>>>((float4*)gy, n4);
    long s4 = ((long)NB * KK * PP) / 4;
    k_zero_f4<<<(int)((s4 + 255) / 256), 256, 0, stream>>>((float4*)xspec, s4);
  }
  // weights -> bf16
  k_f32_to_bf16<<<(16384 + 255) / 256, 256, 0, stream>>>(Bre, Bre_bf, 16384);
  k_f32_to_bf16<<<(16384 + 255) / 256, 256, 0, stream>>>(Bim, Bim_bf, 16384);
  k_f32_to_bf16<<<(49152 + 255) / 256, 256, 0, stream>>>(W1, W1_bf, 49152);
  k_f32_to_bf16<<<(16384 + 255) / 256, 256, 0, stream>>>(W2, W2_bf, 16384);
  k_f32_to_bf16<<<(16384 + 255) / 256, 256, 0, stream>>>(W3, W3_bf, 16384);
  // activations/eigenbasis -> bf16 (+ transposed copies for WMMA operand layouts)
  k_prep<<<(NB * NN + 255) / 256, 256, 0, stream>>>(x_in, va, evecs,
                                                    xin_bf, xaT_bf, evecs_bf, evecsT_bf);
  // spectral projection (split-N atomic reduction)
  dim3 g1((NN + 511) / 512, NB);
  k_gemm1<<<g1, 256, 0, stream>>>(evecsT_bf, xaT_bf, xspec);
  // heat-kernel scaling
  k_scale<<<(NB * KK * PP + 255) / 256, 256, 0, stream>>>(evals, times, xspec, yT_bf);
  // spectral reconstruction
  dim3 g2(((NN / 16) + 7) / 8, NB);
  k_gemm2<<<g2, 256, 0, stream>>>(evecs_bf, yT_bf, xd, xd_bf);
  // sparse gradients
  long st = (long)NB * EE * 32;
  k_spmm<<<(int)((st + 255) / 256), 256, 0, stream>>>(grows, gcols, gxv, gyv, xd, gx, gy);
  // fused gradient features + MLP + residual
  k_tail<<<g2, 256, 0, stream>>>(gx, gy, Bre_bf, Bim_bf, xin_bf, xd_bf,
                                 W1_bf, b1, W2_bf, b2, W3_bf, b3, x_in, out);
}
